// simple_attn_14671608283616
// MI455X (gfx1250) — matrix operand order repaired and checked by reading, not compiled, not run on hardware
//
#include <hip/hip_runtime.h>
#include <hip/hip_bf16.h>
#include <math.h>
#include <stdint.h>

typedef __attribute__((ext_vector_type(16))) __bf16 v16bf;
typedef __attribute__((ext_vector_type(8)))  float  v8f;
typedef __attribute__((ext_vector_type(4)))  unsigned int u32x4;
typedef __attribute__((ext_vector_type(8)))  unsigned int u32x8;

// ---------------------------------------------------------------------------
// Generic WMMA GEMM:  Y[b] = Wt(M,K) * X[b](K,Nn) + bias   (bf16 WMMA, f32 acc)
// Block tile 128x128, BK=32, 256 threads = 8 waves, each wave 4x2 16x16 tiles.
// IN_BF16: X is bf16 (else fp32).  RESID_OUT: write fp32 Y = gemm + resid,
// else write bf16 Y.
// ---------------------------------------------------------------------------
template <bool IN_BF16, bool RESID_OUT>
__global__ __launch_bounds__(256)
void wmma_gemm(const float* __restrict__ Wt, const float* __restrict__ bias,
               const void* __restrict__ Xv, void* __restrict__ Yv,
               const float* __restrict__ resid, int M, int K, int Nn)
{
    constexpr int BM = 128, BN = 128, BK = 32;
    __shared__ __bf16 ldsA[BM * BK];   // [m][k]
    __shared__ __bf16 ldsB[BN * BK];   // [n][k]

    const int b   = blockIdx.z;
    const int m0  = blockIdx.y * BM;
    const int n0  = blockIdx.x * BN;
    const int tid = threadIdx.x;
    const int lane = tid & 31;
    const int wave = tid >> 5;
    const int wm = wave >> 2;          // 0..1  (64 rows each)
    const int wn = wave & 3;           // 0..3  (32 cols each)

    const float*  Xf = (const float*)Xv;
    const __bf16* Xb = (const __bf16*)Xv;
    const char*   Xc = (const char*)Xv;
    const size_t xbase = (size_t)b * K * Nn;
    const size_t esz = IN_BF16 ? 2 : 4;

    v8f acc[4][2] = {};

    for (int k0 = 0; k0 < K; k0 += BK) {
        // ---- stage A tile (weights, fp32 -> bf16), 128x32 row-major
        #pragma unroll
        for (int r = 0; r < 16; ++r) {
            int idx = tid + r * 256;               // 0..4095
            int mm = idx >> 5, kk = idx & 31;
            ldsA[mm * BK + kk] = (__bf16)Wt[(size_t)(m0 + mm) * K + (k0 + kk)];
        }
        // ---- stage B tile (activations), transposed into [n][k]
        #pragma unroll
        for (int r = 0; r < 16; ++r) {
            int idx = tid + r * 256;
            int kk = idx >> 7, nn = idx & 127;
            size_t g = xbase + (size_t)(k0 + kk) * Nn + (n0 + nn);
            float xv = IN_BF16 ? (float)Xb[g] : Xf[g];
            ldsB[nn * BK + kk] = (__bf16)xv;
        }
        // prefetch next K slice of X into caches
        if (k0 + BK < K)
            __builtin_prefetch(Xc + (xbase + (size_t)(k0 + BK) * Nn + n0 + tid) * esz, 0, 1);
        __syncthreads();

        const int koffA = (lane < 16) ? 0 : 8;
        const int rrow  = (lane < 16) ? lane : lane - 16;
        const int koffB = (lane < 16) ? 0 : 16;

        #pragma unroll
        for (int nt = 0; nt < 2; ++nt) {
            union { v16bf v; __bf16 e[16]; } ub;
            const int nb = wn * 32 + nt * 16 + rrow;
            #pragma unroll
            for (int i = 0; i < 16; ++i) ub.e[i] = ldsB[nb * BK + koffA + i + (i & 8)];
            #pragma unroll
            for (int mt = 0; mt < 4; ++mt) {
                union { v16bf v; __bf16 e[16]; } ua;
                const int ma = wm * 64 + mt * 16 + rrow;
                #pragma unroll
                for (int i = 0; i < 8; ++i) {
                    ua.e[i]     = ldsA[ma * BK + koffA + i];
                    ua.e[8 + i] = ldsA[ma * BK + koffA + 16 + i];
                }
                acc[mt][nt] = __builtin_amdgcn_wmma_f32_16x16x32_bf16(
                    false, ua.v, false, ub.v, (short)0, acc[mt][nt], false, false);
            }
        }
        __syncthreads();
    }

    // ---- epilogue: C/D layout: VGPR r -> M = r (lanes 0-15) or r+8 (16-31)
    const int ncol   = n0 + wn * 32 + (lane & 15);
    const int mbase0 = m0 + wm * 64 + ((lane >= 16) ? 8 : 0);
    #pragma unroll
    for (int nt = 0; nt < 2; ++nt) {
        int nc = ncol + nt * 16;
        #pragma unroll
        for (int mt = 0; mt < 4; ++mt) {
            int mb = mbase0 + mt * 16;
            #pragma unroll
            for (int r = 0; r < 8; ++r) {
                int mg = mb + r;
                float v = acc[mt][nt][r] + bias[mg];
                size_t off = (size_t)b * M * Nn + (size_t)mg * Nn + nc;
                if (RESID_OUT) ((float*)Yv)[off] = v + resid[off];
                else           ((__bf16*)Yv)[off] = (__bf16)v;
            }
        }
    }
}

// ---------------------------------------------------------------------------
// Fused RoPE + LayerNorm(k,v) + kv = k^T v / N  (per (b,h)), WMMA accumulate.
// 128 threads = 4 waves; each wave owns a 32-row K sub-chunk; LDS reduce.
// ---------------------------------------------------------------------------
__global__ __launch_bounds__(128)
void kv_wmma_kernel(const __bf16* __restrict__ qkv,
                    const float* __restrict__ kln_w, const float* __restrict__ kln_b,
                    const float* __restrict__ vln_w, const float* __restrict__ vln_b,
                    const float* __restrict__ freqs, float* __restrict__ kv)
{
    constexpr int N = 4096;
    __shared__ __bf16 ldsK[32 * 128];      // A = k^T : [d][n]
    __shared__ __bf16 ldsV[128 * 32];      // B = v   : [n][e]
    __shared__ float  ldsAcc[4 * 4 * 256]; // 4 waves x 4 tiles x (8x32)

    const int b = blockIdx.x >> 3;
    const int h = blockIdx.x & 7;
    const int tid = threadIdx.x;
    const int lane = tid & 31;
    const int wave = tid >> 5;

    const __bf16* kp = qkv + (size_t)(b * 768 + h * 96 + 32) * N;
    const __bf16* vp = qkv + (size_t)(b * 768 + h * 96 + 64) * N;
    const float* f0 = freqs;
    const float* f1 = freqs + 128;

    v8f acc[2][2] = {};

    for (int c0 = 0; c0 < N; c0 += 128) {
        const int n = c0 + tid;
        const float tx = (float)(n & 63);
        const float ty = (float)(n >> 6);
        float kr[32], vr[32];
        #pragma unroll
        for (int p = 0; p < 16; ++p) {
            float ang = tx * f0[h * 16 + p] + ty * f1[h * 16 + p];
            float cs = __cosf(ang), sn = __sinf(ang);
            float k0 = (float)kp[(size_t)(2 * p) * N + n];
            float k1 = (float)kp[(size_t)(2 * p + 1) * N + n];
            float v0 = (float)vp[(size_t)(2 * p) * N + n];
            float v1 = (float)vp[(size_t)(2 * p + 1) * N + n];
            kr[2 * p]     = k0 * cs - k1 * sn;
            kr[2 * p + 1] = k0 * sn + k1 * cs;
            vr[2 * p]     = v0 * cs - v1 * sn;
            vr[2 * p + 1] = v0 * sn + v1 * cs;
        }
        // LayerNorm with unbiased std (ddof=1), eps added to std
        float km = 0.f, vm = 0.f;
        #pragma unroll
        for (int d = 0; d < 32; ++d) { km += kr[d]; vm += vr[d]; }
        km *= (1.f / 32.f); vm *= (1.f / 32.f);
        float ks = 0.f, vs = 0.f;
        #pragma unroll
        for (int d = 0; d < 32; ++d) {
            float a = kr[d] - km; ks += a * a;
            float c = vr[d] - vm; vs += c * c;
        }
        float kinv = 1.f / (sqrtf(ks * (1.f / 31.f)) + 1e-5f);
        float vinv = 1.f / (sqrtf(vs * (1.f / 31.f)) + 1e-5f);
        #pragma unroll
        for (int d = 0; d < 32; ++d) {
            float kn = kln_w[h * 32 + d] * ((kr[d] - km) * kinv) + kln_b[h * 32 + d];
            float vn = vln_w[h * 32 + d] * ((vr[d] - vm) * vinv) + vln_b[h * 32 + d];
            ldsK[d * 128 + tid] = (__bf16)kn;
            ldsV[tid * 32 + d]  = (__bf16)vn;
        }
        __syncthreads();

        const int s = wave;                 // this wave's 32-row K sub-chunk
        const int koffA = (lane < 16) ? 0 : 8;
        const int rrow  = (lane < 16) ? lane : lane - 16;
        const int koffB = (lane < 16) ? 0 : 16;
        #pragma unroll
        for (int nt = 0; nt < 2; ++nt) {
            union { v16bf v; __bf16 e[16]; } ub;
            const int ee = nt * 16 + rrow;
            #pragma unroll
            for (int i = 0; i < 16; ++i) ub.e[i] = ldsV[(s * 32 + koffA + i + (i & 8)) * 32 + ee];
            #pragma unroll
            for (int mt = 0; mt < 2; ++mt) {
                union { v16bf v; __bf16 e[16]; } ua;
                const int dd = mt * 16 + rrow;
                #pragma unroll
                for (int i = 0; i < 8; ++i) {
                    ua.e[i]     = ldsK[dd * 128 + s * 32 + koffA + i];
                    ua.e[8 + i] = ldsK[dd * 128 + s * 32 + koffA + 16 + i];
                }
                acc[mt][nt] = __builtin_amdgcn_wmma_f32_16x16x32_bf16(
                    false, ua.v, false, ub.v, (short)0, acc[mt][nt], false, false);
            }
        }
        __syncthreads();
    }

    // cross-wave reduction in LDS
    #pragma unroll
    for (int mt = 0; mt < 2; ++mt)
        #pragma unroll
        for (int nt = 0; nt < 2; ++nt)
            #pragma unroll
            for (int r = 0; r < 8; ++r)
                ldsAcc[(wave * 4 + mt * 2 + nt) * 256 + r * 32 + lane] = acc[mt][nt][r];
    __syncthreads();

    for (int i = tid; i < 1024; i += 128) {
        int d = i >> 5, e = i & 31;
        int mt = d >> 4, nt = e >> 4;
        int dd = d & 15, ee = e & 15;
        int rr = dd & 7;
        int ln = ((dd >= 8) ? 16 : 0) + ee;
        float sum = 0.f;
        #pragma unroll
        for (int w = 0; w < 4; ++w)
            sum += ldsAcc[(w * 4 + mt * 2 + nt) * 256 + rr * 32 + ln];
        kv[((size_t)(b * 8 + h) * 32 + d) * 32 + e] = sum * (1.f / 4096.f);
    }
}

// out = q @ kv + x   (kv is tiny & L2-resident; q loads coalesced along n)
__global__ __launch_bounds__(256)
void attn_out_kernel(const __bf16* __restrict__ qkv, const float* __restrict__ kvm,
                     const float* __restrict__ x, float* __restrict__ ret)
{
    constexpr int N = 4096;
    size_t idx = (size_t)blockIdx.x * 256 + threadIdx.x;
    int n = idx & 4095;
    int c = (int)((idx >> 12) & 255);
    int b = (int)(idx >> 20);
    int h = c >> 5, e = c & 31;
    const __bf16* qp = qkv + (size_t)(b * 768 + h * 96) * N + n;
    const float* kp = kvm + (size_t)(b * 8 + h) * 1024 + e;
    float acc = x[idx];
    #pragma unroll
    for (int d = 0; d < 32; ++d)
        acc += (float)qp[(size_t)d * N] * kp[d * 32];
    ret[idx] = acc;
}

// ---------------------------------------------------------------------------
// Depthwise 3x3 SAME + bias + exact GELU.
// One block per (b,c) plane: the whole 64x64 bf16 plane (8 KB) is DMA'd into
// LDS by the Tensor Data Mover (1-D D#: tile_dim0 = tensor_dim0 = 4096 elems,
// data_size = 2B, workgroup_mask = 0 since we are not in a cluster), then the
// 9-tap stencil reads LDS with 9x reuse.
// ---------------------------------------------------------------------------
__global__ __launch_bounds__(256)
void dwconv_gelu_kernel(const __bf16* __restrict__ z1, const float* __restrict__ dw_w,
                        const float* __restrict__ dw_b, __bf16* __restrict__ z2)
{
    __shared__ __bf16 plane[4096];
    const int bc  = blockIdx.x;            // b*256 + c
    const int c   = bc & 255;
    const int tid = threadIdx.x;

    if (tid < 32) {                        // wave 0 issues the TDM descriptor
        unsigned lds_off = (unsigned)(unsigned long long)(uintptr_t)&plane[0];
        unsigned long long ga =
            (unsigned long long)(uintptr_t)(z1 + (size_t)bc * 4096);
        u32x4 g0;
        g0[0] = 1u;                                        // count=1, user D#
        g0[1] = lds_off;                                   // lds_addr
        g0[2] = (unsigned)ga;                              // global_addr[31:0]
        g0[3] = (unsigned)((ga >> 32) & 0x01ffffffu)       // global_addr[56:32]
                | (2u << 30);                              // type = 2 (image)
        u32x8 g1;
        g1[0] = 0x00010000u;   // workgroup_mask=0, data_size=1 (2 bytes)
        g1[1] = 0x10000000u;   // tensor_dim0[15:0]=4096 in bits[63:48]
        g1[2] = 0x00010000u;   // tensor_dim0[31:16]=0, tensor_dim1=1
        g1[3] = 0x10000000u;   // tile_dim0 = 4096 in bits[127:112]
        g1[4] = 0u;            // tile_dim1=0 (unused), tile_dim2=0
        g1[5] = 4096u;         // tensor_dim0_stride[31:0]
        g1[6] = 0u;            // tensor_dim0_stride[47:32], dim1_stride lo
        g1[7] = 0u;
        asm volatile("tensor_load_to_lds %0, %1" :: "s"(g0), "s"(g1) : "memory");
        __builtin_amdgcn_s_wait_tensorcnt(0);
    }
    __syncthreads();

    float w[9];
    #pragma unroll
    for (int i = 0; i < 9; ++i) w[i] = dw_w[c * 9 + i];
    const float bb = dw_b[c];

    for (int i = tid; i < 4096; i += 256) {
        int xx = i & 63, yy = i >> 6;
        float acc = bb;
        #pragma unroll
        for (int dy = -1; dy <= 1; ++dy) {
            int y2 = yy + dy; if (y2 < 0 || y2 > 63) continue;
            #pragma unroll
            for (int dx = -1; dx <= 1; ++dx) {
                int x2 = xx + dx; if (x2 < 0 || x2 > 63) continue;
                acc += (float)plane[y2 * 64 + x2] * w[(dy + 1) * 3 + (dx + 1)];
            }
        }
        float g = 0.5f * acc * (1.f + erff(acc * 0.70710678118654752f));
        z2[(size_t)bc * 4096 + i] = (__bf16)g;
    }
}

extern "C" void kernel_launch(void* const* d_in, const int* in_sizes, int n_in,
                              void* d_out, int out_size, void* d_ws, size_t ws_size,
                              hipStream_t stream)
{
    const float* x     = (const float*)d_in[0];
    const float* qkv_w = (const float*)d_in[1];
    const float* qkv_b = (const float*)d_in[2];
    const float* o1_w  = (const float*)d_in[3];
    const float* o1_b  = (const float*)d_in[4];
    const float* o2_w  = (const float*)d_in[5];
    const float* o2_b  = (const float*)d_in[6];
    const float* dw_w  = (const float*)d_in[7];
    const float* dw_b  = (const float*)d_in[8];
    const float* kln_w = (const float*)d_in[9];
    const float* kln_b = (const float*)d_in[10];
    const float* vln_w = (const float*)d_in[11];
    const float* vln_b = (const float*)d_in[12];
    const float* freqs = (const float*)d_in[13];

    const int B = 16, C = 256, N = 4096;

    // workspace layout (bytes):
    //   [0, 100663296)            qkv bf16 (B,768,N)   -- later reused by z1/z2
    //   [100663296, 167772160)    ret fp32 (B,C,N)
    //   [167772160, 168296448)    kv  fp32 (B,8,32,32)
    char* ws = (char*)d_ws;
    __bf16* qkv = (__bf16*)ws;
    float*  ret = (float*)(ws + 100663296);
    float*  kvm = (float*)(ws + 167772160);
    __bf16* z1  = (__bf16*)ws;                 // reuse qkv region after attention
    __bf16* z2  = (__bf16*)(ws + 33554432);

    // 1) qkv = qkv_w @ x + qkv_b          (bf16 WMMA GEMM, fp32 in, bf16 out)
    wmma_gemm<false, false><<<dim3(N / 128, 768 / 128, B), 256, 0, stream>>>(
        qkv_w, qkv_b, (const void*)x, (void*)qkv, nullptr, 768, C, N);

    // 2) kv[b,h] = LN(rope(k))^T LN(rope(v)) / N   (fused, WMMA accumulate)
    kv_wmma_kernel<<<B * 8, 128, 0, stream>>>(qkv, kln_w, kln_b, vln_w, vln_b,
                                              freqs, kvm);

    // 3) ret = q @ kv + x
    attn_out_kernel<<<(B * C * N) / 256, 256, 0, stream>>>(qkv, kvm, x, ret);

    // 4) z1 = o1_w @ ret + o1_b           (fp32 in, bf16 out)
    wmma_gemm<false, false><<<dim3(N / 128, C / 128, B), 256, 0, stream>>>(
        o1_w, o1_b, (const void*)ret, (void*)z1, nullptr, C, C, N);

    // 5) z2 = gelu(dwconv3x3(z1) + dw_b)  (TDM plane load -> LDS stencil)
    dwconv_gelu_kernel<<<B * C, 256, 0, stream>>>(z1, dw_w, dw_b, z2);

    // 6) out = o2_w @ z2 + o2_b + x       (bf16 in, fp32 out, residual = original x)
    wmma_gemm<true, true><<<dim3(N / 128, C / 128, B), 256, 0, stream>>>(
        o2_w, o2_b, (const void*)z2, d_out, x, C, C, N);
}
